// MLPMoE_old_62156766708055
// MI455X (gfx1250) — compile-verified, matrix operand order checked by reading, not run on hardware
//
#include <hip/hip_runtime.h>
#include <hip/hip_bf16.h>

#define NEXP  8
#define HID   256
#define MTILE 256
#define ROWH  264   // halves per activation row (528 B: 16B-aligned, bank-step 4)

typedef __attribute__((ext_vector_type(16))) _Float16 v16h;
typedef __attribute__((ext_vector_type(8)))  float    v8f;

union FragAB { uint4 u[2]; v16h h; };

// ---------------------------------------------------------------------------
// Pre-pass: convert W1/W2 (f32 [E][256][256]) into f16 packed WMMA B-fragment
// layout in d_ws. uint index = ((layer*8 + e)*128 + tile)*256 + lane*8 + i,
// tile = kt*16 + nt; each uint packs halves (K=k, K=k+1) for that lane/vgpr.
// ---------------------------------------------------------------------------
__global__ __launch_bounds__(256)
void pack_w_kernel(const float* __restrict__ W1, const float* __restrict__ W2,
                   unsigned int* __restrict__ wpk)
{
  const int idx   = blockIdx.x * 256 + threadIdx.x;   // [0, 524288)
  const int layer = idx >> 18;
  const int r     = idx & 0x3FFFF;
  const int e     = r >> 15;
  const int t     = (r >> 8) & 127;
  const int l     = (r >> 3) & 31;
  const int i     = r & 7;
  const int kt = t >> 4, nt = t & 15;
  const int klocal = ((l & 16) ? 8 : 0) + ((i < 4) ? (2 * i) : (16 + 2 * (i - 4)));
  const int k = kt * 32 + klocal;
  const int n = nt * 16 + (l & 15);
  const float* W = layer ? W2 : W1;
  const int base = e * HID * HID + k * HID + n;
  _Float16 lo = (_Float16)W[base];
  _Float16 hi = (_Float16)W[base + HID];
  unsigned int u = (unsigned int)__builtin_bit_cast(unsigned short, lo)
                 | ((unsigned int)__builtin_bit_cast(unsigned short, hi) << 16);
  wpk[idx] = u;
}

// Issue one 64KB stage-unit (async global->LDS) for unit u into buf (u&1).
__device__ __forceinline__ void issue_stage(const unsigned int* __restrict__ wpk,
                                            int u, unsigned wst_lds, int tid)
{
  const int e  = u >> 2;
  const int L  = (u >> 1) & 1;
  const int ph = u & 1;
  const unsigned int* wbase = wpk + (((L << 3) + e) << 15);
  const unsigned dbase = wst_lds + ((unsigned)(u & 1) << 16);   // 65536 B buffers
  #pragma unroll
  for (int it = 0; it < 16; ++it) {
    int q = tid + (it << 8);               // dst uint4 index 0..4095
    int tileLocal = q >> 6;
    int kt  = tileLocal >> 3;
    int ntl = tileLocal & 7;
    unsigned dst    = dbase + ((unsigned)q << 4);
    unsigned srcoff = (unsigned)((((kt*16 + ph*8 + ntl) << 6) + (q & 63)) << 4);
    asm volatile("global_load_async_to_lds_b128 %0, %1, %2"
                 :: "v"(dst), "v"(srcoff), "s"(wbase) : "memory");
  }
}

// ---------------------------------------------------------------------------
// Fused MoE-MLP, 256 points/block, 8 experts looped in LDS.
// Two 16-row strips per wave (2x B reuse per LDS read, dual WMMA chains);
// double-buffered async weight staging pipelined one unit ahead of the GEMM.
// ---------------------------------------------------------------------------
__global__ __launch_bounds__(256)
void moe_occ_kernel(const float* __restrict__ coords,
                    const float* __restrict__ W0, const float* __restrict__ b0,
                    const float* __restrict__ b1, const float* __restrict__ b2,
                    const float* __restrict__ W3, const float* __restrict__ b3,
                    const unsigned int* __restrict__ wpk,
                    float* __restrict__ out, int N)
{
  __shared__ unsigned short act[MTILE * ROWH];  // 135168 B f16 activations
  __shared__ unsigned int   wst[2 * 16384];     // 131072 B double-buffered frags
  __shared__ float          cs[MTILE * 4];      //   4096 B coords tile
  __shared__ float          fs[1024];           //   4096 B W0/b0 (L0), W3 (L3)
  __shared__ float          bstd[2][HID];       //   2048 B bias, layer-parity buffered

  const int tid  = threadIdx.x;
  const int lane = tid & 31;
  const int wave = tid >> 5;
  const int p0   = blockIdx.x * MTILE;

  const unsigned wst_lds = (unsigned)(unsigned long long)(void*)wst;

  // kick off the staging pipeline (unit 0 -> buf 0); overlaps with layer 0
  issue_stage(wpk, 0, wst_lds, tid);

  {
    int p = p0 + tid;
    float c0 = 0.f, c1 = 0.f, c2 = 0.f;
    if (p < N) { c0 = coords[3*p]; c1 = coords[3*p+1]; c2 = coords[3*p+2]; }
    cs[4*tid] = c0; cs[4*tid+1] = c1; cs[4*tid+2] = c2; cs[4*tid+3] = 0.f;
  }

  float rmax = -3.402823466e38f;

  for (int e = 0; e < NEXP; ++e) {
    // ----- layer 0 (K=3): stage W0/b0 in LDS, thread owns one row -----
    #pragma unroll
    for (int it = 0; it < 4; ++it) {
      int i = tid + (it << 8);
      fs[i] = (i < 768) ? W0[e*768 + i] : b0[e*HID + (i - 768)];
    }
    __syncthreads();
    {
      const float c0 = cs[4*tid], c1 = cs[4*tid+1], c2 = cs[4*tid+2];
      unsigned int* rowu = (unsigned int*)(act + tid * ROWH);
      for (int np = 0; np < 128; ++np) {
        const int n0 = 2*np, n1 = n0 + 1;
        float x0 = fmaf(c2, fs[512+n0], fmaf(c1, fs[256+n0], fmaf(c0, fs[n0], fs[768+n0])));
        float x1 = fmaf(c2, fs[512+n1], fmaf(c1, fs[256+n1], fmaf(c0, fs[n1], fs[768+n1])));
        x0 = fmaxf(x0, 0.f); x1 = fmaxf(x1, 0.f);
        unsigned int u = (unsigned int)__builtin_bit_cast(unsigned short, (_Float16)x0)
                       | ((unsigned int)__builtin_bit_cast(unsigned short, (_Float16)x1) << 16);
        rowu[np] = u;
      }
    }
    __syncthreads();

    // ----- layers 1 & 2: 256x256 @ 256x256 via WMMA, two strips per wave -----
    for (int L = 0; L < 2; ++L) {
      const float* bias = (L == 0 ? b1 : b2) + e * HID;

      // Hold A fragments (full K) for BOTH strips -> safe in-place update.
      v16h a0[8], a1[8];
      {
        const int koff = (lane & 16) ? 1 : 0;
        const uint4* rp0 = (const uint4*)(act + ((wave << 4)       + (lane & 15)) * ROWH);
        const uint4* rp1 = (const uint4*)(act + (((wave + 8) << 4) + (lane & 15)) * ROWH);
        #pragma unroll
        for (int kt = 0; kt < 8; ++kt) {
          FragAB fa;
          fa.u[0] = rp0[kt*4 + koff]; fa.u[1] = rp0[kt*4 + koff + 2]; a0[kt] = fa.h;
          fa.u[0] = rp1[kt*4 + koff]; fa.u[1] = rp1[kt*4 + koff + 2]; a1[kt] = fa.h;
        }
      }

      for (int ph = 0; ph < 2; ++ph) {
        const int u = ((((e << 1) | L) << 1) | ph);
        // unit u's async loads (issued last iteration) must be complete
        asm volatile("s_wait_asynccnt 0x0" ::: "memory");
        if (ph == 0) bstd[L][tid] = bias[tid];  // layer-parity buffer: race-free
        __syncthreads();
        // prefetch next stage-unit; overlaps fully with this GEMM phase
        if (u + 1 < 4 * NEXP) issue_stage(wpk, u + 1, wst_lds, tid);

        const uint4* wst4 = ((const uint4*)wst) + ((u & 1) << 12);
        for (int ntl = 0; ntl < 8; ++ntl) {
          v8f acc0 = {0.f,0.f,0.f,0.f,0.f,0.f,0.f,0.f};
          v8f acc1 = {0.f,0.f,0.f,0.f,0.f,0.f,0.f,0.f};
          const int boff = (ntl << 6) + (lane << 1);  // u4 index; kt stride = 512 u4
          FragAB fb, fbn;
          fb.u[0] = wst4[boff]; fb.u[1] = wst4[boff + 1];
          #pragma unroll
          for (int kt = 0; kt < 8; ++kt) {
            if (kt < 7) {   // prefetch next B fragment while WMMA pair runs
              fbn.u[0] = wst4[boff + (kt+1)*512];
              fbn.u[1] = wst4[boff + (kt+1)*512 + 1];
            }
            acc0 = __builtin_amdgcn_wmma_f32_16x16x32_f16(
                     false, a0[kt], false, fb.h, (short)0, acc0, false, false);
            acc1 = __builtin_amdgcn_wmma_f32_16x16x32_f16(
                     false, a1[kt], false, fb.h, (short)0, acc1, false, false);
            if (kt < 7) fb = fbn;
          }
          // epilogue: bias + ReLU + f16, in-place into both strips
          const int nc = (ph << 7) + (ntl << 4) + (lane & 15);
          const float bb = bstd[L][nc];
          const int mb0 = (wave << 4)       + ((lane & 16) ? 8 : 0);
          const int mb1 = ((wave + 8) << 4) + ((lane & 16) ? 8 : 0);
          #pragma unroll
          for (int v = 0; v < 8; ++v) {
            float x0 = fmaxf(acc0[v] + bb, 0.f);
            float x1 = fmaxf(acc1[v] + bb, 0.f);
            act[(mb0 + v)*ROWH + nc] = __builtin_bit_cast(unsigned short, (_Float16)x0);
            act[(mb1 + v)*ROWH + nc] = __builtin_bit_cast(unsigned short, (_Float16)x1);
          }
        }
        // no end barrier: next unit's begin-barrier orders buf reuse
      }
    }

    // ----- layer 3 (256 -> 1) + running max; thread owns one row -----
    __syncthreads();           // all waves' layer-2 strip writes visible
    fs[tid] = W3[e*HID + tid];
    __syncthreads();
    {
      float s = b3[e];
      const unsigned short* row = act + tid * ROWH;
      for (int k = 0; k < HID; ++k)
        s = fmaf((float)__builtin_bit_cast(_Float16, row[k]), fs[k], s);
      rmax = fmaxf(rmax, s);
    }
    __syncthreads();           // protect act/fs before next expert overwrites
  }

  {
    int p = p0 + tid;
    if (p < N) out[p] = rmax;
  }
}

extern "C" void kernel_launch(void* const* d_in, const int* in_sizes, int n_in,
                              void* d_out, int out_size, void* d_ws, size_t ws_size,
                              hipStream_t stream)
{
  const float* coords = (const float*)d_in[0];
  const float* W0 = (const float*)d_in[1];
  const float* b0 = (const float*)d_in[2];
  const float* W1 = (const float*)d_in[3];
  const float* b1 = (const float*)d_in[4];
  const float* W2 = (const float*)d_in[5];
  const float* b2 = (const float*)d_in[6];
  const float* W3 = (const float*)d_in[7];
  const float* b3 = (const float*)d_in[8];
  float* out = (float*)d_out;
  const int N = in_sizes[0] / 3;
  unsigned int* wpk = (unsigned int*)d_ws;   // needs 2 MB scratch

  pack_w_kernel<<<2048, 256, 0, stream>>>(W1, W2, wpk);

  const int nblk = (N + MTILE - 1) / MTILE;
  moe_occ_kernel<<<nblk, 256, 0, stream>>>(coords, W0, b0, b1, b2, W3, b3,
                                           wpk, out, N);
}
